// GTEProgramClassification_27986006900873
// MI455X (gfx1250) — matrix operand order repaired and checked by reading, not compiled, not run on hardware
//
#include <hip/hip_runtime.h>
#include <hip/hip_bf16.h>

// ---------------------------------------------------------------------------
// GTE program classification: embed-gather -> ragged masked GRU -> LayerNorm
// -> linear classifier.  MI455X (gfx1250) wave32 + bf16 WMMA implementation.
//
// One wave = one 16-node tile (M=16 matches v_wmma_f32_16x16x32_bf16).
// A fragments gathered straight into contiguous-k A layout (k = 16*half + e) from a bf16 embed table
// (prep kernel converts f32 -> bf16 into d_ws). B fragments read from padded
// row-major bf16 weight copies in LDS. Gate math / masking done in the C/D
// accumulator layout (lane = gate column, VGPR = node row).
// ---------------------------------------------------------------------------

typedef __attribute__((ext_vector_type(4)))  __bf16 v4bf;
typedef __attribute__((ext_vector_type(8)))  __bf16 v8bf;
typedef __attribute__((ext_vector_type(16))) __bf16 v16bf;
typedef __attribute__((ext_vector_type(8)))  float  v8f;

#define HD     128      // hidden dim D
#define G3     384      // 3*D gate width
#define NCLS   104      // classes
#define LMAX   8        // mailbox length
#define ROWP   136      // bf16 row pitch (128 + 8 pad -> 272B, 4-dword bank stride)
#define HPITCH 132      // f32 h row pitch (528B)

// LDS layout (bytes)
#define OFF_WIH   0
#define OFF_WHH   104448                 // 384*136*2
#define OFF_FCW   208896                 // +384*136*2
#define OFF_BIAS  239360                 // +112*136*2
#define OFF_LNG   242432                 // +768*4
#define OFF_LNB   242944                 // +128*4
#define OFF_FCB   243456                 // +128*4
#define OFF_WAVE  243904                 // +112*4 (rounded)
#define WAVE_SZ   12864                  // 16*132*4 + 16*136*2 + 64
#define LDS_TOTAL (OFF_WAVE + 4*WAVE_SZ) // 295360 B < 320KB

#define WMMA_BF16(A, B, C) \
  __builtin_amdgcn_wmma_f32_16x16x32_bf16(false, (A), false, (B), (short)0, (C), false, false)

__device__ __forceinline__ v16bf load_a_frag(const __bf16* p /*16B aligned*/) {
  // A fragment per lane: K offsets {0..15} relative to p.
  v8bf lo = *(const v8bf*)(p);
  v8bf hi = *(const v8bf*)(p + 8);
  v16bf r;
#pragma unroll
  for (int i = 0; i < 8; ++i) { r[i] = lo[i]; r[i + 8] = hi[i]; }
  return r;
}

__device__ __forceinline__ float sigmoidf_(float x) {
  return 1.0f / (1.0f + __expf(-x));
}

// ---------------------------------------------------------------------------
__global__ void prep_embed_bf16(const float* __restrict__ src,
                                __bf16* __restrict__ dst, int n) {
  int i = (blockIdx.x * blockDim.x + threadIdx.x) * 4;
  if (i + 3 < n) {
    float4 v = *(const float4*)(src + i);
    v4bf o;
    o[0] = (__bf16)v.x; o[1] = (__bf16)v.y; o[2] = (__bf16)v.z; o[3] = (__bf16)v.w;
    *(v4bf*)(dst + i) = o;
  } else {
    for (; i < n; ++i) dst[i] = (__bf16)src[i];
  }
}

// ---------------------------------------------------------------------------
__global__ void __launch_bounds__(128, 1)
gte_gru_wmma(const int* __restrict__ token_id,
             const int* __restrict__ mailbox_idx,
             const int* __restrict__ lengths,
             const float* __restrict__ embed,
             const float* __restrict__ W_ih,
             const float* __restrict__ W_hh,
             const float* __restrict__ b_ih,
             const float* __restrict__ b_hh,
             const float* __restrict__ ln_g,
             const float* __restrict__ ln_b,
             const float* __restrict__ fc_w,
             const float* __restrict__ fc_b,
             const __bf16* __restrict__ embed_bf,
             float* __restrict__ out,
             int n_dst) {
  extern __shared__ char smem[];
  __bf16* wih  = (__bf16*)(smem + OFF_WIH);
  __bf16* whh  = (__bf16*)(smem + OFF_WHH);
  __bf16* fcw  = (__bf16*)(smem + OFF_FCW);
  float*  bsl  = (float*)(smem + OFF_BIAS);   // [768] = b_ih | b_hh
  float*  lngl = (float*)(smem + OFF_LNG);
  float*  lnbl = (float*)(smem + OFF_LNB);
  float*  fcbl = (float*)(smem + OFF_FCB);

  const int lane = threadIdx.x & 31;
  const int wave = threadIdx.x >> 5;
  const int hlf  = lane >> 4;     // half-wave: selects K sub-offsets / M high rows
  const int mrow = lane & 15;     // A row (node-in-tile) and C/D column index

  char*   wbase = smem + OFF_WAVE + wave * WAVE_SZ;
  float*  hf = (float*)wbase;                     // [16][132] f32 master state
  __bf16* hb = (__bf16*)(wbase + 16 * HPITCH * 4);// [16][136] bf16 state for WMMA A
  int*    nst = (int*)(wbase + 16 * HPITCH * 4 + 16 * ROWP * 2); // [16] steps

  // ---- stage weights/biases (bf16, padded rows) ---------------------------
  for (int i = threadIdx.x; i < G3 * HD; i += blockDim.x) {
    int r = i >> 7, k = i & 127;
    wih[r * ROWP + k] = (__bf16)W_ih[i];
    whh[r * ROWP + k] = (__bf16)W_hh[i];
  }
  for (int i = threadIdx.x; i < 112 * HD; i += blockDim.x) {
    int r = i >> 7, k = i & 127;
    fcw[r * ROWP + k] = (r < NCLS) ? (__bf16)fc_w[r * HD + k] : (__bf16)0.0f;
  }
  for (int i = threadIdx.x; i < G3; i += blockDim.x) {
    bsl[i] = b_ih[i]; bsl[G3 + i] = b_hh[i];
  }
  for (int i = threadIdx.x; i < HD; i += blockDim.x) {
    lngl[i] = ln_g[i]; lnbl[i] = ln_b[i];
  }
  for (int i = threadIdx.x; i < 112; i += blockDim.x)
    fcbl[i] = (i < NCLS) ? fc_b[i] : 0.0f;

  // ---- per-tile setup -----------------------------------------------------
  const int tile0 = (blockIdx.x * 4 + wave) * 16;
  int nodeA = tile0 + mrow; if (nodeA >= n_dst) nodeA = n_dst - 1;  // clamp, guard stores
  const int lenA = lengths[nodeA];
  if (hlf == 0) nst[mrow] = lenA - 1;

  // h0 = embed[token at slot len-1], f32 + bf16 copies in LDS
  for (int nd = 0; nd < 16; ++nd) {
    int nn = tile0 + nd; if (nn >= n_dst) nn = n_dst - 1;
    int tk = token_id[mailbox_idx[nn * LMAX + (lengths[nn] - 1)]];
    const float* src = embed + (long)tk * HD;
    for (int d = lane; d < HD; d += 32) {
      float v = src[d];
      hf[nd * HPITCH + d] = v;
      hb[nd * ROWP + d] = (__bf16)v;
    }
  }
  __syncthreads();

  int nstv[8];
#pragma unroll
  for (int v = 0; v < 8; ++v) nstv[v] = nst[v + hlf * 8];

  // ---- masked ragged GRU: 7 fixed steps -----------------------------------
  for (int t = 0; t < LMAX - 1; ++t) {
    int tk = token_id[mailbox_idx[nodeA * LMAX + t]];
    const __bf16* xrow = embed_bf + (long)tk * HD;
    if (t + 1 < LMAX - 1) {
      int tk2 = token_id[mailbox_idx[nodeA * LMAX + t + 1]];
      __builtin_prefetch(embed_bf + (long)tk2 * HD, 0, 1);
    }

    v16bf ax[4], ah[4];
#pragma unroll
    for (int c = 0; c < 4; ++c) {
      int k0 = c * 32 + hlf * 16;
      ax[c] = load_a_frag(xrow + k0);
      ah[c] = load_a_frag(hb + mrow * ROWP + k0);
    }

    for (int j = 0; j < 8; ++j) {  // 8 column-tiles of each 128-wide gate
      v8f gir = {0,0,0,0,0,0,0,0}, giz = gir, gin = gir;
      v8f ghr = gir, ghz = gir, ghn = gir;
#pragma unroll
      for (int c = 0; c < 4; ++c) {
        int bofs = c * 32 + hlf * 16;
        int rr = (0   + j * 16 + mrow) * ROWP + bofs;
        int rz = (128 + j * 16 + mrow) * ROWP + bofs;
        int rn = (256 + j * 16 + mrow) * ROWP + bofs;
        v16bf bir_ = *(const v16bf*)(wih + rr);
        v16bf biz_ = *(const v16bf*)(wih + rz);
        v16bf bin_ = *(const v16bf*)(wih + rn);
        gir = WMMA_BF16(ax[c], bir_, gir);
        giz = WMMA_BF16(ax[c], biz_, giz);
        gin = WMMA_BF16(ax[c], bin_, gin);
        v16bf bhr_ = *(const v16bf*)(whh + rr);
        v16bf bhz_ = *(const v16bf*)(whh + rz);
        v16bf bhn_ = *(const v16bf*)(whh + rn);
        ghr = WMMA_BF16(ah[c], bhr_, ghr);
        ghz = WMMA_BF16(ah[c], bhz_, ghz);
        ghn = WMMA_BF16(ah[c], bhn_, ghn);
      }
      int col = j * 16 + mrow;                 // gate column (C/D lane role)
      float bir = bsl[col],        biz = bsl[128 + col], bin = bsl[256 + col];
      float bhr = bsl[384 + col],  bhz = bsl[512 + col], bhn = bsl[640 + col];
#pragma unroll
      for (int v = 0; v < 8; ++v) {
        int M = v + hlf * 8;                   // node row (C/D VGPR role)
        float r  = sigmoidf_(gir[v] + ghr[v] + bir + bhr);
        float z  = sigmoidf_(giz[v] + ghz[v] + biz + bhz);
        float nn = tanhf(gin[v] + bin + r * (ghn[v] + bhn));
        float ho = hf[M * HPITCH + col];
        float hn = (t < nstv[v]) ? ((1.0f - z) * nn + z * ho) : ho;
        hf[M * HPITCH + col] = hn;
        hb[M * ROWP + col] = (__bf16)hn;
      }
    }
    // LDS ops from one wave stay in order; state is wave-private -> no barrier
  }

  // ---- LayerNorm (bypass for len==1) + classifier WMMA --------------------
  float sum = 0.0f, sq = 0.0f;
  for (int d = 0; d < HD; d += 4) {
    float4 v = *(const float4*)(hf + mrow * HPITCH + d);
    sum += v.x + v.y + v.z + v.w;
    sq  += v.x * v.x + v.y * v.y + v.z * v.z + v.w * v.w;
  }
  float mu   = sum * (1.0f / HD);
  float var  = sq * (1.0f / HD) - mu * mu;
  float rstd = rsqrtf(var + 1e-5f);
  bool bypass = (lenA == 1);

  v16bf af[4];
#pragma unroll
  for (int c = 0; c < 4; ++c) {
#pragma unroll
    for (int e = 0; e < 16; ++e) {
      int k = c * 32 + hlf * 16 + e;
      float hv = hf[mrow * HPITCH + k];
      float lv = (hv - mu) * rstd * lngl[k] + lnbl[k];
      af[c][e] = (__bf16)(bypass ? hv : lv);
    }
  }

  for (int jt = 0; jt < 7; ++jt) {
    v8f acc = {0,0,0,0,0,0,0,0};
#pragma unroll
    for (int c = 0; c < 4; ++c) {
      v16bf b = *(const v16bf*)(fcw + (jt * 16 + mrow) * ROWP + c * 32 + hlf * 16);
      acc = WMMA_BF16(af[c], b, acc);
    }
    int col = jt * 16 + mrow;
    if (col < NCLS) {
      float bias = fcbl[col];
#pragma unroll
      for (int v = 0; v < 8; ++v) {
        int nodeM = tile0 + v + hlf * 8;
        if (nodeM < n_dst) out[(long)nodeM * NCLS + col] = acc[v] + bias;
      }
    }
  }
}

// ---------------------------------------------------------------------------
extern "C" void kernel_launch(void* const* d_in, const int* in_sizes, int n_in,
                              void* d_out, int out_size, void* d_ws, size_t ws_size,
                              hipStream_t stream) {
  const int*   token_id = (const int*)d_in[0];
  const int*   mailbox  = (const int*)d_in[1];
  const int*   lengths  = (const int*)d_in[2];
  const float* embed    = (const float*)d_in[3];
  const float* W_ih     = (const float*)d_in[4];
  const float* W_hh     = (const float*)d_in[5];
  const float* b_ih     = (const float*)d_in[6];
  const float* b_hh     = (const float*)d_in[7];
  const float* ln_g     = (const float*)d_in[8];
  const float* ln_b     = (const float*)d_in[9];
  const float* fc_w     = (const float*)d_in[10];
  const float* fc_b     = (const float*)d_in[11];
  float* out = (float*)d_out;

  const int embed_elems = in_sizes[3];   // V*D
  const int n_dst       = in_sizes[2];   // N_DST

  __bf16* embed_bf = (__bf16*)d_ws;      // bf16 embed table (V*D*2 bytes)

  // 1) convert embeddings to bf16 once per launch (L2-resident table)
  int thr = 256;
  int nquads = (embed_elems + 3) / 4;
  prep_embed_bf16<<<(nquads + thr - 1) / thr, thr, 0, stream>>>(
      embed, embed_bf, embed_elems);

  // 2) fused gather + GRU + LayerNorm + classifier
  int blocks = (n_dst + 63) / 64;        // 4 waves/block, 16 nodes/wave
  gte_gru_wmma<<<blocks, 128, LDS_TOTAL, stream>>>(
      token_id, mailbox, lengths, embed, W_ih, W_hh, b_ih, b_hh,
      ln_g, ln_b, fc_w, fc_b, embed_bf, out, n_dst);
}
